// Quantize_12240656794057
// MI455X (gfx1250) — matrix operand order repaired and checked by reading, not compiled, not run on hardware
//
#include <hip/hip_runtime.h>
#include <hip/hip_bf16.h>
#include <math.h>
#include <stdint.h>

#define DIM    256
#define NEMB   2048
#define NROWS  65536      // 16*64*64

typedef __attribute__((ext_vector_type(16))) __bf16          v16bf;
typedef __attribute__((ext_vector_type(8)))  float           v8f;
typedef __attribute__((ext_vector_type(4)))  float           v4f;
typedef __attribute__((ext_vector_type(8)))  unsigned short  u16x8;
typedef __attribute__((ext_vector_type(16))) unsigned short  u16x16;

// ---- d_out layout (tuple flattened in return order) ----
static constexpr size_t OFF_DIFF = (size_t)NROWS * DIM;        // 16777216
static constexpr size_t OFF_IND  = OFF_DIFF + 1;               // 16777217
static constexpr size_t OFF_PERP = OFF_IND + NROWS;            // 16842753

// ---- ws layout (bytes) ----
static constexpr size_t WS_EHI  = 0;                                  // [2048][256] bf16 hi
static constexpr size_t WS_ELO  = WS_EHI  + (size_t)NEMB * DIM * 2;   // [2048][256] bf16 lo
static constexpr size_t WS_EF32 = WS_ELO  + (size_t)NEMB * DIM * 2;   // [2048][256] f32 (exact, transposed)
static constexpr size_t WS_ENRM = WS_EF32 + (size_t)NEMB * DIM * 4;   // [2048] ||e||^2
static constexpr size_t WS_CNT  = WS_ENRM + (size_t)NEMB * 4;         // [2048] int counts
static constexpr size_t WS_IND  = WS_CNT  + (size_t)NEMB * 4;         // [65536] int indices
static constexpr size_t WS_PART = WS_IND  + (size_t)NROWS * 4;        // [65536] partial diff sums

__device__ __forceinline__ void split_bf16(float x, unsigned short& hi, unsigned short& lo) {
    unsigned ub = __float_as_uint(x);
    float hf = __uint_as_float(ub & 0xFFFF0000u);   // exact bf16 prefix (truncate)
    float lf = x - hf;                              // exact residual in f32
    hi = (unsigned short)(ub >> 16);
    lo = (unsigned short)(__float_as_uint(lf) >> 16);
}

// ---------------- K0: transpose/split codebook, norms, zero counts ----------------
__global__ __launch_bounds__(256)
void vq_prep(const float* __restrict__ embed,          // [DIM][NEMB]
             unsigned short* __restrict__ ehi,
             unsigned short* __restrict__ elo,
             float* __restrict__ ef32,
             float* __restrict__ enrm,
             int* __restrict__ counts) {
    const int n = blockIdx.x * blockDim.x + threadIdx.x;
    if (n >= NEMB) return;
    counts[n] = 0;
    float nrm = 0.f;
    for (int k = 0; k < DIM; ++k) {
        float x = embed[(size_t)k * NEMB + n];       // coalesced across n
        nrm += x * x;
        unsigned short h, l;
        split_bf16(x, h, l);
        ehi[(size_t)n * DIM + k] = h;
        elo[(size_t)n * DIM + k] = l;
        ef32[(size_t)n * DIM + k] = x;
    }
    enrm[n] = nrm;
}

// ---------------- K1: WMMA argmin over codes ----------------
// Block = 256 threads = 8 waves; wave w owns 32 rows (two 16-row A panels).
__global__ __launch_bounds__(256)
void vq_argmin(const float* __restrict__ input,
               const unsigned short* __restrict__ ehi,
               const unsigned short* __restrict__ elo,
               const float* __restrict__ enrm,
               int* __restrict__ ind,
               float* __restrict__ out_ind) {
    const int lane = threadIdx.x & 31;
    const int wave = threadIdx.x >> 5;
    const int sel  = lane >> 4;          // which 16-lane half
    const int m    = lane & 15;
    const int row0 = (blockIdx.x * 8 + wave) * 32;

    // ---- Stage ||e||^2 table (8KB) into LDS via CDNA5 async load-to-LDS ----
    __shared__ float s_enrm[NEMB];
    {
        const unsigned lds0 = (unsigned)(uintptr_t)(&s_enrm[0]);
        const int tid = (int)threadIdx.x;
        const float* g0 = enrm + tid * 4;            // 4 floats per lane (b128)
        const unsigned l0 = lds0 + tid * 16;
        asm volatile("global_load_async_to_lds_b128 %0, %1, off"
                     :: "v"(l0), "v"(g0) : "memory");
        const float* g1 = g0 + 1024;
        const unsigned l1 = l0 + 4096;
        asm volatile("global_load_async_to_lds_b128 %0, %1, off"
                     :: "v"(l1), "v"(g1) : "memory");
        asm volatile("s_wait_asynccnt 0" ::: "memory");
        __syncthreads();
    }

    // ---- Load & split two 16x256 A panels into bf16 hi/lo fragments ----
    v16bf a_hi[2][8], a_lo[2][8];
#pragma unroll
    for (int p = 0; p < 2; ++p) {
        const float* arow = input + (size_t)(row0 + p * 16 + m) * DIM;
#pragma unroll
        for (int c = 0; c < 8; ++c) {
            const int kb = c * 32;
            u16x16 uh, ul;
#pragma unroll
            for (int half = 0; half < 2; ++half) {
                // A layout: lanes<16 take K {0..15}+kb, lanes>=16 take K {16..31}+kb
                const int koff = kb + sel * 16 + half * 8;
                const v4f p0 = *(const v4f*)(arow + koff);
                const v4f p1 = *(const v4f*)(arow + koff + 4);
                const int e = half * 8;
#pragma unroll
                for (int j = 0; j < 4; ++j) {
                    unsigned short h, l;
                    split_bf16(p0[j], h, l); uh[e + j] = h;     ul[e + j] = l;
                    split_bf16(p1[j], h, l); uh[e + 4 + j] = h; ul[e + 4 + j] = l;
                }
            }
            a_hi[p][c] = __builtin_bit_cast(v16bf, uh);
            a_lo[p][c] = __builtin_bit_cast(v16bf, ul);
        }
    }

    float minv[2][8];
    int   mini[2][8];
#pragma unroll
    for (int p = 0; p < 2; ++p)
#pragma unroll
        for (int r = 0; r < 8; ++r) { minv[p][r] = 3.4e38f; mini[p][r] = 0; }

    // ---- Sweep all 2048 codes in 16-wide tiles; B reused by both row panels ----
    for (int t = 0; t < NEMB / 16; ++t) {
        const int n = t * 16 + m;                     // this lane's column
        const unsigned short* bh = ehi + (size_t)n * DIM + sel * 16;
        const unsigned short* bl = elo + (size_t)n * DIM + sel * 16;

        if (t + 1 < NEMB / 16)
            __builtin_prefetch(ehi + (size_t)(n + 16) * DIM, 0, 1);

        v8f acc0 = {0.f, 0.f, 0.f, 0.f, 0.f, 0.f, 0.f, 0.f};
        v8f acc1 = {0.f, 0.f, 0.f, 0.f, 0.f, 0.f, 0.f, 0.f};
#pragma unroll
        for (int c = 0; c < 8; ++c) {
            // B layout: lane = column, element i = K (c*32 + sel*16 + i); contiguous in embT
            u16x8 h0 = *(const u16x8*)(bh + c * 32);
            u16x8 h1 = *(const u16x8*)(bh + c * 32 + 8);
            u16x8 l0 = *(const u16x8*)(bl + c * 32);
            u16x8 l1 = *(const u16x8*)(bl + c * 32 + 8);
            v16bf Bh = __builtin_bit_cast(v16bf, __builtin_shufflevector(
                h0, h1, 0,1,2,3,4,5,6,7,8,9,10,11,12,13,14,15));
            v16bf Bl = __builtin_bit_cast(v16bf, __builtin_shufflevector(
                l0, l1, 0,1,2,3,4,5,6,7,8,9,10,11,12,13,14,15));
            // dot ~= hi*hi + hi*lo + lo*hi (f32 acc); alternate acc0/acc1 to break RAW chains
            acc0 = __builtin_amdgcn_wmma_f32_16x16x32_bf16(false, a_hi[0][c], false, Bh,
                                                           (short)0, acc0, false, false);
            acc1 = __builtin_amdgcn_wmma_f32_16x16x32_bf16(false, a_hi[1][c], false, Bh,
                                                           (short)0, acc1, false, false);
            acc0 = __builtin_amdgcn_wmma_f32_16x16x32_bf16(false, a_hi[0][c], false, Bl,
                                                           (short)0, acc0, false, false);
            acc1 = __builtin_amdgcn_wmma_f32_16x16x32_bf16(false, a_hi[1][c], false, Bl,
                                                           (short)0, acc1, false, false);
            acc0 = __builtin_amdgcn_wmma_f32_16x16x32_bf16(false, a_lo[0][c], false, Bh,
                                                           (short)0, acc0, false, false);
            acc1 = __builtin_amdgcn_wmma_f32_16x16x32_bf16(false, a_lo[1][c], false, Bh,
                                                           (short)0, acc1, false, false);
        }

        const float en = s_enrm[n];
#pragma unroll
        for (int r = 0; r < 8; ++r) {
            // argmin of (||e||^2 - 2 x.e); ||x||^2 is row-constant, dropped
            float d0 = en - 2.0f * acc0[r];
            if (d0 < minv[0][r]) { minv[0][r] = d0; mini[0][r] = n; }
            float d1 = en - 2.0f * acc1[r];
            if (d1 < minv[1][r]) { minv[1][r] = d1; mini[1][r] = n; }
        }
    }

    // ---- Reduce across the 16 lanes holding the same output row ----
#pragma unroll
    for (int p = 0; p < 2; ++p) {
#pragma unroll
        for (int r = 0; r < 8; ++r) {
            float v = minv[p][r];
            int   i = mini[p][r];
#pragma unroll
            for (int s = 1; s < 16; s <<= 1) {
                float ov = __shfl_xor(v, s, 32);
                int   oi = __shfl_xor(i, s, 32);
                if (ov < v || (ov == v && oi < i)) { v = ov; i = oi; }  // first-index tiebreak
            }
            if (m == 0) {
                const int row = row0 + p * 16 + r + 8 * sel;
                ind[row] = i;
                out_ind[row] = (float)i;
            }
        }
    }
}

// ---------------- K2: gather codes, straight-through output, diff partials, histogram ----------------
__global__ __launch_bounds__(256)
void vq_gather(const float* __restrict__ input,
               const float* __restrict__ ef32,       // [2048][256] exact codes
               const int* __restrict__ ind,
               float* __restrict__ out_q,
               int* __restrict__ counts,
               float* __restrict__ part) {
    __shared__ float sdata[256];
    const int row = blockIdx.x;
    const int k   = threadIdx.x;
    const int idx = ind[row];
    const float x = input[(size_t)row * DIM + k];
    const float q = ef32[(size_t)idx * DIM + k];
    out_q[(size_t)row * DIM + k] = x + (q - x);      // straight-through estimator
    const float d = q - x;
    sdata[k] = d * d;
    __syncthreads();
    for (int s = 128; s > 0; s >>= 1) {
        if (k < s) sdata[k] += sdata[k + s];
        __syncthreads();
    }
    if (k == 0) {
        part[row] = sdata[0];
        atomicAdd(&counts[idx], 1);                  // integer atomic: deterministic
    }
}

// ---------------- K3: final diff + perplexity ----------------
__global__ __launch_bounds__(1024)
void vq_final(const float* __restrict__ part,
              const int* __restrict__ counts,
              float* __restrict__ d_out) {
    __shared__ float s1[1024];
    __shared__ float s2[1024];
    const int t = threadIdx.x;
    float acc = 0.f;
    for (int i = t; i < NROWS; i += 1024) acc += part[i];
    float ent = 0.f;
    for (int j = t; j < NEMB; j += 1024) {
        int c = counts[j];
        if (c > 0) {                                  // c==0 contributes 0 (clip semantics)
            float p = (float)c * (1.0f / (float)NROWS);
            ent += p * logf(p);
        }
    }
    s1[t] = acc; s2[t] = ent;
    __syncthreads();
    for (int s = 512; s > 0; s >>= 1) {
        if (t < s) { s1[t] += s1[t + s]; s2[t] += s2[t + s]; }
        __syncthreads();
    }
    if (t == 0) {
        d_out[OFF_DIFF] = s1[0] / (float)((size_t)NROWS * DIM);
        d_out[OFF_PERP] = expf(-s2[0]);
    }
}

extern "C" void kernel_launch(void* const* d_in, const int* in_sizes, int n_in,
                              void* d_out, int out_size, void* d_ws, size_t ws_size,
                              hipStream_t stream) {
    (void)in_sizes; (void)n_in; (void)out_size; (void)ws_size;
    const float* input = (const float*)d_in[0];      // [16,64,64,256] f32
    const float* embed = (const float*)d_in[1];      // [256,2048] f32
    float* out = (float*)d_out;

    char* ws = (char*)d_ws;
    unsigned short* ehi  = (unsigned short*)(ws + WS_EHI);
    unsigned short* elo  = (unsigned short*)(ws + WS_ELO);
    float*          ef32 = (float*)(ws + WS_EF32);
    float*          enrm = (float*)(ws + WS_ENRM);
    int*            cnt  = (int*)(ws + WS_CNT);
    int*            ind  = (int*)(ws + WS_IND);
    float*          part = (float*)(ws + WS_PART);

    float* out_q   = out;
    float* out_ind = out + OFF_IND;

    vq_prep  <<<NEMB / 256, 256, 0, stream>>>(embed, ehi, elo, ef32, enrm, cnt);
    vq_argmin<<<NROWS / 256, 256, 0, stream>>>(input, ehi, elo, enrm, ind, out_ind);
    vq_gather<<<NROWS, 256, 0, stream>>>(input, ef32, ind, out_q, cnt, part);
    vq_final <<<1, 1024, 0, stream>>>(part, cnt, out);
}
